// F3Layer_85177791414398
// MI455X (gfx1250) — compile-verified, matrix operand order checked by reading, not run on hardware
//
#include <hip/hip_runtime.h>

// ---------------------------------------------------------------------------
// F3Layer for MI455X (gfx1250, wave32, WMMA + TDM).
//   - A converted once to fp16 (32MB, L2-resident for all passes)
//   - L@hops eliminated: L@Hk = Hk - Hk+1 (compute H3 instead of 3 L-GEMMs)
//   - adjacency mask derived from A16 > 0 (adj_mask / L inputs never read)
//   - hop GEMM: 1-wave blocks, K split 4 ways (1024 WGs/hop), B operand
//     staged to LDS by the Tensor Data Mover (double buffered,
//     s_wait_tensorcnt pipelining); deterministic partial-sum reduction
//   - masked low-rank flash attention with v_wmma_f32_16x16x32_f16
// ---------------------------------------------------------------------------

#define NN     4096
#define DIN    128
#define SUBD   32
#define NK     3
#define KSPLIT 4
#define KC     (NN / KSPLIT)       // 1024 K per block
#define LDS_PITCH_H 40             // halves per LDS row (64B data + 16B pad)

typedef __attribute__((ext_vector_type(8)))  _Float16 v8h;
typedef __attribute__((ext_vector_type(16))) _Float16 v16h;
typedef __attribute__((ext_vector_type(8)))  float    v8f;
typedef __attribute__((ext_vector_type(4)))  unsigned int v4u;
typedef __attribute__((ext_vector_type(8)))  int      v8i;
typedef __attribute__((ext_vector_type(4)))  int      v4i;

// Row-major fp16 fragment loader (global). CDNA5 16-bit A/B layout:
// lanes 0-15 hold K {0..7,16..23}, lanes 16-31 hold K {8..15,24..31}.
__device__ __forceinline__ v16h load_frag16(const _Float16* __restrict__ base,
                                            int row0, int ld, int k0) {
  const int lane = threadIdx.x & 31;
  const _Float16* p = base + (size_t)(row0 + (lane & 15)) * (size_t)ld
                           + (size_t)(k0 + ((lane >> 4) << 3));
  v8h lo = *(const v8h*)(p);
  v8h hi = *(const v8h*)(p + 16);
  v16h out;
#pragma unroll
  for (int e = 0; e < 8; ++e) { out[e] = lo[e]; out[8 + e] = hi[e]; }
  return out;
}

__device__ __forceinline__ v8f wmma16(v16h a, v16h b, v8f c) {
  return __builtin_amdgcn_wmma_f32_16x16x32_f16(false, a, false, b,
                                                (short)0, c, false, false);
}

__device__ __forceinline__ float hmax16(float v) {
#pragma unroll
  for (int d = 1; d < 16; d <<= 1) v = fmaxf(v, __shfl_xor(v, d, 16));
  return v;
}
__device__ __forceinline__ float hsum16(float v) {
#pragma unroll
  for (int d = 1; d < 16; d <<= 1) v += __shfl_xor(v, d, 16);
  return v;
}

// TDM: async-load a 128-row x 64B 2D tile (row stride 8192B in memory) into
// LDS with +16B per-row padding (LDS pitch 80B). D# per ISA 08_async_tensor §8.
// This toolchain's builtin is the 6-arg form:
//   (uint32x4 g0, int32x8 g1, int32x4 g2, int32x4 g3, int32x8 g4, i32 cpol)
__device__ __forceinline__ void tdm_load_slab(unsigned int lds_off,
                                              const _Float16* gsrc) {
  unsigned long long ga = (unsigned long long)(size_t)gsrc;
  v4u g0;
  g0[0] = 1u;                                       // count=1, user descriptor
  g0[1] = lds_off;                                  // LDS byte address
  g0[2] = (unsigned int)ga;                         // global_addr[31:0]
  g0[3] = (unsigned int)((ga >> 32) & 0x01FFFFFFu)  // global_addr[56:32]
          | (2u << 30);                             // type=2 ("image")
  v8i g1;
  g1[0] = (int)((2u << 16)      // data_size = 4B units
              | (1u << 20)      // pad_enable
              | (3u << 22)      // pad_interval code 3 = 16 DWORDs (64B)
              | (3u << 25));    // pad_amount  code 3 = 4 DWORDs (16B)
  g1[1] = (int)((2048u & 0xFFFFu) << 16);                    // tensor_dim0 lo16
  g1[2] = (int)(((2048u >> 16) & 0xFFFFu) | (128u << 16));   // dim0 hi | dim1 lo
  g1[3] = (int)(0u | (16u << 16));                           // dim1 hi | tile_dim0=16 dwords
  g1[4] = (int)128;                                          // tile_dim1=128 rows, tile_dim2=0
  g1[5] = (int)2048;                                         // tensor_dim0_stride = 2048 dwords
  g1[6] = 0;
  g1[7] = 0;
  v4i gz4 = {0, 0, 0, 0};
  v8i gz8 = {0, 0, 0, 0, 0, 0, 0, 0};
  __builtin_amdgcn_tensor_load_to_lds(g0, g1, gz4, gz4, gz8, 0);
}

// ---------------- conversion / transpose ----------------------------------
__global__ __launch_bounds__(256) void cvt_a_kernel(const float4* __restrict__ src,
                                                    _Float16* __restrict__ dst) {
  size_t i = (size_t)blockIdx.x * 256 + threadIdx.x;   // over NN*NN/4
  float4 v = src[i];
  _Float16* d = dst + 4 * i;
  d[0] = (_Float16)v.x; d[1] = (_Float16)v.y;
  d[2] = (_Float16)v.z; d[3] = (_Float16)v.w;
}

// [NN][DIN] f32 -> [DIN][NN] f16
__global__ __launch_bounds__(256) void transpose_h_kernel(const float* __restrict__ src,
                                                          _Float16* __restrict__ dst) {
  int idx = blockIdx.x * 256 + threadIdx.x;            // over NN*DIN
  int row = idx >> 7, col = idx & 127;
  dst[(size_t)col * NN + row] = (_Float16)src[idx];
}

// ---------------- hop GEMM partial: one wave = 16 rows x 128 cols, K chunk --
// B slab (128 x 32 f16 per K-step) DMA'd into LDS by the TDM, double buffered.
__global__ __launch_bounds__(32)
void hop_gemm_tdm_kernel(const _Float16* __restrict__ A16,
                         const _Float16* __restrict__ BT16,
                         float* __restrict__ Cpart) {   // [KSPLIT][NN][DIN]
  const int lane = threadIdx.x & 31;
  const int i0   = blockIdx.x * 16;
  const int kbeg = blockIdx.y * KC;

  __shared__ __align__(16) _Float16 Bs[2][128 * LDS_PITCH_H];
  const unsigned int lds0 = (unsigned int)(size_t)(&Bs[0][0]);
  const unsigned int lds1 = (unsigned int)(size_t)(&Bs[1][0]);

  v8f acc[8];
#pragma unroll
  for (int t = 0; t < 8; ++t) acc[t] = v8f{};

  tdm_load_slab(lds0, BT16 + kbeg);                  // prime buffer 0

  const int nsteps = KC / 32;
  for (int s = 0; s < nsteps; ++s) {
    const int k0 = kbeg + s * 32;
    v16h a = load_frag16(A16, i0, NN, k0);           // issue A load early
    if (s + 1 < nsteps) {                            // prefetch next slab
      tdm_load_slab((s & 1) ? lds0 : lds1, BT16 + k0 + 32);
      __builtin_amdgcn_s_wait_tensorcnt(1);          // current slab landed
    } else {
      __builtin_amdgcn_s_wait_tensorcnt(0);
    }
    const _Float16* bs = (s & 1) ? &Bs[1][0] : &Bs[0][0];
#pragma unroll
    for (int t = 0; t < 8; ++t) {
      const _Float16* p = bs + (t * 16 + (lane & 15)) * LDS_PITCH_H
                             + ((lane >> 4) << 3);
      v8h lo = *(const v8h*)(p);
      v8h hi = *(const v8h*)(p + 16);
      v16h b;
#pragma unroll
      for (int e = 0; e < 8; ++e) { b[e] = lo[e]; b[8 + e] = hi[e]; }
      acc[t] = wmma16(a, b, acc[t]);
    }
  }

  const int n  = lane & 15;
  const int mb = (lane >> 4) << 3;
  float* Cp = Cpart + (size_t)blockIdx.y * NN * DIN;
#pragma unroll
  for (int t = 0; t < 8; ++t) {
#pragma unroll
    for (int g = 0; g < 8; ++g) {
      Cp[(size_t)(i0 + mb + g) * DIN + t * 16 + n] = acc[t][g];
    }
  }
}

// ---------------- deterministic K-partial reduction + transpose/f16 --------
__global__ __launch_bounds__(256)
void reduce_hop_kernel(const float* __restrict__ Cpart,
                       float* __restrict__ Hk, _Float16* __restrict__ HkT) {
  int idx = blockIdx.x * 256 + threadIdx.x;          // over NN*DIN
  float s = 0.f;
#pragma unroll
  for (int p = 0; p < KSPLIT; ++p) s += Cpart[(size_t)p * NN * DIN + idx];
  Hk[idx] = s;
  int row = idx >> 7, col = idx & 127;
  if (HkT) HkT[(size_t)col * NN + row] = (_Float16)s;
}

// ---------------- Zt[n][s] = sum_c Hk[n][c] * Uk[c][s]  (+f16 copies) -------
__global__ __launch_bounds__(128)
void z_kernel(const float* __restrict__ Hk, const float* __restrict__ Uk,
              float* __restrict__ Zt, _Float16* __restrict__ Zt16,
              _Float16* __restrict__ Z16) {
  __shared__ float Us[DIN * SUBD];
  for (int i = threadIdx.x; i < DIN * SUBD; i += 128) Us[i] = Uk[i];
  __syncthreads();
  int s  = threadIdx.x & 31;
  int ii = threadIdx.x >> 5;
  int row = blockIdx.x * 4 + ii;
  const float* h = Hk + (size_t)row * DIN;
  float acc = 0.f;
#pragma unroll 8
  for (int c = 0; c < DIN; ++c) acc += h[c] * Us[c * SUBD + s];
  Zt[(size_t)row * SUBD + s]   = acc;
  Zt16[(size_t)row * SUBD + s] = (_Float16)acc;
  Z16[(size_t)s * NN + row]    = (_Float16)acc;
}

// ---------------- Gram partials: Mpart[k][chunk] = Zt_chunk^T Zt_chunk ------
__global__ __launch_bounds__(1024)
void m_part_kernel(const float* __restrict__ Zt, float* __restrict__ Mpart) {
  int k = blockIdx.x, chunk = blockIdx.y;            // 8 chunks of 512 rows
  const float* Z = Zt + (size_t)k * NN * SUBD + (size_t)chunk * 512 * SUBD;
  __shared__ float tile[256 * SUBD];
  int i = threadIdx.x >> 5, j = threadIdx.x & 31;
  float acc = 0.f;
  for (int n0 = 0; n0 < 512; n0 += 256) {
    __syncthreads();
    for (int t = threadIdx.x; t < 256 * SUBD; t += 1024)
      tile[t] = Z[(size_t)n0 * SUBD + t];
    __syncthreads();
#pragma unroll 4
    for (int n = 0; n < 256; ++n) acc += tile[n * SUBD + i] * tile[n * SUBD + j];
  }
  Mpart[((size_t)(k * 8 + chunk)) * 1024 + i * 32 + j] = acc;
}

// ---------------- 32x32 SPD inverse + logdet (Gauss-Jordan, 1 wave / k) -----
// Reduces the 8 Gram partials in fixed order (deterministic), forms
// M = I + coeff*Gram, inverts, returns logdet/2.
__global__ __launch_bounds__(32)
void inv_kernel(const float* __restrict__ Mpart, float* __restrict__ Minv,
                float* __restrict__ crs, float coeff) {
  int k = blockIdx.x;
  int r = threadIdx.x;
  float row[64];
  for (int j = 0; j < 32; ++j) {
    float acc = 0.f;
#pragma unroll
    for (int p = 0; p < 8; ++p)
      acc += Mpart[((size_t)(k * 8 + p)) * 1024 + r * 32 + j];
    row[j] = (j == r ? 1.f : 0.f) + coeff * acc;
  }
  for (int j = 0; j < 32; ++j) row[32 + j] = (j == r) ? 1.f : 0.f;
  __shared__ float prow[64];
  __shared__ float logdet;
  if (r == 0) logdet = 0.f;
  for (int p = 0; p < 32; ++p) {
    __syncthreads();
    if (r == p) {
      float piv = row[p];
      logdet += __logf(piv);
      float inv = 1.0f / piv;
      for (int j = 0; j < 64; ++j) { row[j] *= inv; prow[j] = row[j]; }
    }
    __syncthreads();
    if (r != p) {
      float f = row[p];
      for (int j = 0; j < 64; ++j) row[j] -= f * prow[j];
    }
  }
  __syncthreads();
  for (int j = 0; j < 32; ++j) Minv[(size_t)k * 1024 + r * 32 + j] = row[32 + j];
  if (r == 0) crs[k] = 0.5f * logdet;
}

// ---------------- weights = softmax(crs); laps = softplus(lambda) -----------
__global__ void wts_kernel(const float* __restrict__ crs,
                           const float* __restrict__ lambda_laps,
                           float* __restrict__ wts, float* __restrict__ laps,
                           float* __restrict__ out_tail) {
  if (threadIdx.x == 0) {
    float m = fmaxf(crs[0], fmaxf(crs[1], crs[2]));
    float e0 = __expf(crs[0] - m), e1 = __expf(crs[1] - m), e2 = __expf(crs[2] - m);
    float inv = 1.0f / (e0 + e1 + e2);
    float w0 = e0 * inv, w1 = e1 * inv, w2 = e2 * inv;
    wts[0] = w0; wts[1] = w1; wts[2] = w2;
    out_tail[0] = w0; out_tail[1] = w1; out_tail[2] = w2;
    for (int k = 0; k < NK; ++k) laps[k] = log1pf(__expf(lambda_laps[k]));
  }
}

// ---------------- Wt16 = (Zt @ Minv) in fp16 --------------------------------
__global__ __launch_bounds__(256)
void wt_kernel(const float* __restrict__ Zt, const float* __restrict__ Minv,
               _Float16* __restrict__ Wt16) {
  int k = blockIdx.y;
  __shared__ float Ms[1024];
  for (int t = threadIdx.x; t < 1024; t += 256) Ms[t] = Minv[(size_t)k * 1024 + t];
  __syncthreads();
  int idx = blockIdx.x * 256 + threadIdx.x;          // n*32 + j over NN*SUBD
  int nrow = idx >> 5, j = idx & 31;
  const float* z = Zt + (size_t)k * NN * SUBD + (size_t)nrow * SUBD;
  float acc = 0.f;
#pragma unroll
  for (int s = 0; s < 32; ++s) acc += z[s] * Ms[s * 32 + j];
  Wt16[(size_t)k * NN * SUBD + idx] = (_Float16)acc;
}

// ---------------- G[k][c][j] = sum_s U[k][c][s] * Minv[k][s][j] -------------
__global__ __launch_bounds__(256)
void g_kernel(const float* __restrict__ U, const float* __restrict__ Minv,
              float* __restrict__ G) {
  int idx = blockIdx.x * 256 + threadIdx.x;
  if (idx >= NK * DIN * SUBD) return;
  int k = idx / (DIN * SUBD);
  int rem = idx - k * DIN * SUBD;
  int c = rem >> 5, j = rem & 31;
  const float* u = U + (size_t)k * DIN * SUBD + (size_t)c * SUBD;
  const float* m = Minv + (size_t)k * 1024;
  float acc = 0.f;
#pragma unroll
  for (int s = 0; s < 32; ++s) acc += u[s] * m[s * 32 + j];
  G[idx] = acc;
}

// ---------------- flash attention: O = softmax(Zt.Wt^T + mask) @ Zt ---------
__global__ __launch_bounds__(64)
void attn_kernel(const _Float16* __restrict__ Zt16,
                 const _Float16* __restrict__ Wt16,
                 const _Float16* __restrict__ Z16,
                 const _Float16* __restrict__ A16,
                 float* __restrict__ O) {
  const int k    = blockIdx.y;
  const int wave = threadIdx.x >> 5;
  const int lane = threadIdx.x & 31;
  const int i0   = blockIdx.x * 32 + wave * 16;
  const _Float16* Q = Zt16 + (size_t)k * NN * SUBD;
  const _Float16* W = Wt16 + (size_t)k * NN * SUBD;
  const _Float16* V = Z16  + (size_t)k * SUBD * NN;

  __shared__ __align__(16) _Float16 Pall[2][16 * 32];
  _Float16* Plds = Pall[wave];

  v16h q = load_frag16(Q, i0, SUBD, 0);               // 16 rows x 32 dims
  v8f o0 = v8f{}, o1 = v8f{};
  float mrun[8], lrun[8];
#pragma unroll
  for (int g = 0; g < 8; ++g) { mrun[g] = -1e30f; lrun[g] = 0.f; }

  const int nloc = lane & 15;
  const int mb   = (lane >> 4) << 3;

  for (int j0 = 0; j0 < NN; j0 += 32) {
    // issue all global loads up front; latency hides behind WMMA + softmax
    v16h b0 = load_frag16(W, j0,      SUBD, 0);
    v16h b1 = load_frag16(W, j0 + 16, SUBD, 0);
    float a0m[8], a1m[8];
#pragma unroll
    for (int g = 0; g < 8; ++g) {
      const int irow = i0 + mb + g;
      a0m[g] = (float)A16[(size_t)irow * NN + j0 + nloc];
      a1m[g] = (float)A16[(size_t)irow * NN + j0 + 16 + nloc];
    }
    v16h v0 = load_frag16(V, 0,  NN, j0);             // V dims 0..15
    v16h v1 = load_frag16(V, 16, NN, j0);             // V dims 16..31

    v8f s0 = v8f{}, s1 = v8f{};
    s0 = wmma16(q, b0, s0);
    s1 = wmma16(q, b1, s1);

#pragma unroll
    for (int g = 0; g < 8; ++g) {
      float f0 = s0[g] + (a0m[g] == 0.f ? -1e9f : 0.f);
      float f1 = s1[g] + (a1m[g] == 0.f ? -1e9f : 0.f);
      float t  = hmax16(fmaxf(f0, f1));
      float mnew  = fmaxf(mrun[g], t);
      float scale = __expf(mrun[g] - mnew);
      float p0 = __expf(f0 - mnew);
      float p1 = __expf(f1 - mnew);
      float ps = hsum16(p0 + p1);
      lrun[g] = lrun[g] * scale + ps;
      mrun[g] = mnew;
      o0[g] *= scale;
      o1[g] *= scale;
      Plds[(mb + g) * 32 + nloc]      = (_Float16)p0;
      Plds[(mb + g) * 32 + 16 + nloc] = (_Float16)p1;
    }
    // C-layout -> A-layout transpose of P through wave-private LDS
    {
      const _Float16* pp = Plds + (lane & 15) * 32 + ((lane >> 4) << 3);
      v8h lo = *(const v8h*)(pp);
      v8h hi = *(const v8h*)(pp + 16);
      v16h pf;
#pragma unroll
      for (int e = 0; e < 8; ++e) { pf[e] = lo[e]; pf[8 + e] = hi[e]; }
      o0 = wmma16(pf, v0, o0);
      o1 = wmma16(pf, v1, o1);
    }
  }

#pragma unroll
  for (int g = 0; g < 8; ++g) {
    float inv = 1.0f / lrun[g];
    int irow = i0 + mb + g;
    float* Ok = O + ((size_t)k * NN + irow) * SUBD;
    Ok[nloc]      = o0[g] * inv;
    Ok[16 + nloc] = o1[g] * inv;
  }
}

// ---------------- fused epilogue: agg + lap + shrink + LayerNorm ------------
__global__ __launch_bounds__(128)
void final_kernel(const float* __restrict__ H,
                  const float* __restrict__ H1, const float* __restrict__ H2,
                  const float* __restrict__ H3,
                  const float* __restrict__ O,   // [3][NN][32]
                  const float* __restrict__ G,   // [3][128][32]
                  const float* __restrict__ wts, const float* __restrict__ laps,
                  const float* __restrict__ threshold,
                  const float* __restrict__ gamma_, const float* __restrict__ beta_,
                  float* __restrict__ out) {
  const int i = blockIdx.x;
  const int c = threadIdx.x;
  __shared__ float Orow[NK * SUBD];
  __shared__ float w_s[NK], l_s[NK];
  if (c < NK * SUBD) Orow[c] = O[(size_t)(c >> 5) * NN * SUBD + (size_t)i * SUBD + (c & 31)];
  if (c < NK) { w_s[c] = wts[c]; l_s[c] = laps[c]; }
  __syncthreads();

  float agg = 0.f;
#pragma unroll
  for (int k = 0; k < NK; ++k) {
    const float* g = G + (size_t)k * DIN * SUBD + (size_t)c * SUBD;
    const float* o = Orow + k * SUBD;
    float s = 0.f;
#pragma unroll
    for (int j = 0; j < SUBD; ++j) s += o[j] * g[j];
    agg += w_s[k] * s;
  }
  size_t off = (size_t)i * DIN + c;
  float h0 = H[off], h1 = H1[off], h2 = H2[off], h3 = H3[off];
  float lap = l_s[0] * (h0 - h1) + l_s[1] * (h1 - h2) + l_s[2] * (h2 - h3);
  float hh = h0 + 0.5f * agg - 0.5f * lap;            // ETA = 0.5
  float mag = fabsf(hh) - threshold[c];
  float v = fmaxf(mag, 0.f) * (hh >= 0.f ? 1.f : -1.f);

  __shared__ float rbuf[DIN];
  rbuf[c] = v;
  __syncthreads();
  for (int s2 = 64; s2 > 0; s2 >>= 1) {
    if (c < s2) rbuf[c] += rbuf[c + s2];
    __syncthreads();
  }
  float mu = rbuf[0] * (1.0f / DIN);
  __syncthreads();
  float d = v - mu;
  rbuf[c] = d * d;
  __syncthreads();
  for (int s2 = 64; s2 > 0; s2 >>= 1) {
    if (c < s2) rbuf[c] += rbuf[c + s2];
    __syncthreads();
  }
  float var = rbuf[0] * (1.0f / DIN);
  out[off] = d * rsqrtf(var + 1e-5f) * gamma_[c] + beta_[c];
}

// ---------------------------------------------------------------------------
extern "C" void kernel_launch(void* const* d_in, const int* in_sizes, int n_in,
                              void* d_out, int out_size, void* d_ws, size_t ws_size,
                              hipStream_t stream) {
  (void)in_sizes; (void)n_in; (void)out_size; (void)ws_size;
  const float* H        = (const float*)d_in[0];
  const float* A        = (const float*)d_in[1];
  // d_in[2] (adj_mask) and d_in[3] (L) are algebraically redundant — unused.
  const float* U        = (const float*)d_in[4];
  const float* lam      = (const float*)d_in[5];
  const float* thres    = (const float*)d_in[6];
  const float* ln_gamma = (const float*)d_in[7];
  const float* ln_beta  = (const float*)d_in[8];
  float* out = (float*)d_out;

  char* w = (char*)d_ws;
  size_t off = 0;
  auto alloc = [&](size_t bytes) -> char* {
    char* p = w + off;
    off += (bytes + 255) & ~(size_t)255;
    return p;
  };
  _Float16* A16   = (_Float16*)alloc((size_t)NN * NN * 2);   // 32 MB, L2-resident
  _Float16* H0T   = (_Float16*)alloc((size_t)DIN * NN * 2);
  _Float16* H1T   = (_Float16*)alloc((size_t)DIN * NN * 2);
  _Float16* H2T   = (_Float16*)alloc((size_t)DIN * NN * 2);
  float*    H1    = (float*)alloc((size_t)NN * DIN * 4);
  float*    H2    = (float*)alloc((size_t)NN * DIN * 4);
  float*    H3    = (float*)alloc((size_t)NN * DIN * 4);
  float*    Cpart = (float*)alloc((size_t)KSPLIT * NN * DIN * 4);
  float*    Zt    = (float*)alloc((size_t)NK * NN * SUBD * 4);
  _Float16* Zt16  = (_Float16*)alloc((size_t)NK * NN * SUBD * 2);
  _Float16* Z16   = (_Float16*)alloc((size_t)NK * SUBD * NN * 2);
  _Float16* Wt16  = (_Float16*)alloc((size_t)NK * NN * SUBD * 2);
  float*    Mpart = (float*)alloc((size_t)NK * 8 * 1024 * 4);
  float*    Minv  = (float*)alloc((size_t)NK * 1024 * 4);
  float*    crs   = (float*)alloc(256);
  float*    wts   = (float*)alloc(256);
  float*    laps  = (float*)alloc(256);
  float*    G     = (float*)alloc((size_t)NK * DIN * SUBD * 4);
  float*    O     = (float*)alloc((size_t)NK * NN * SUBD * 4);

  const float coeff = (float)SUBD / ((float)NN * 0.25f);     // sub/(n*eps^2)

  // 1) fp16 operator + transposed H0
  cvt_a_kernel<<<(NN * (size_t)NN / 4) / 256, 256, 0, stream>>>((const float4*)A, A16);
  transpose_h_kernel<<<(NN * DIN) / 256, 256, 0, stream>>>(H, H0T);

  // 2) hops via TDM-fed WMMA GEMM, K split 4 ways, deterministic reduction
  dim3 ggrid(NN / 16, KSPLIT);
  hop_gemm_tdm_kernel<<<ggrid, 32, 0, stream>>>(A16, H0T, Cpart);
  reduce_hop_kernel<<<(NN * DIN) / 256, 256, 0, stream>>>(Cpart, H1, H1T);
  hop_gemm_tdm_kernel<<<ggrid, 32, 0, stream>>>(A16, H1T, Cpart);
  reduce_hop_kernel<<<(NN * DIN) / 256, 256, 0, stream>>>(Cpart, H2, H2T);
  hop_gemm_tdm_kernel<<<ggrid, 32, 0, stream>>>(A16, H2T, Cpart);
  reduce_hop_kernel<<<(NN * DIN) / 256, 256, 0, stream>>>(Cpart, H3, (_Float16*)nullptr);

  // 3) subspace projections Zt[k] = Hk @ U[k]
  z_kernel<<<NN / 4, 128, 0, stream>>>(H,  U,                       Zt,
                                       Zt16, Z16);
  z_kernel<<<NN / 4, 128, 0, stream>>>(H1, U + (size_t)DIN * SUBD,  Zt + (size_t)NN * SUBD,
                                       Zt16 + (size_t)NN * SUBD,    Z16 + (size_t)SUBD * NN);
  z_kernel<<<NN / 4, 128, 0, stream>>>(H2, U + 2 * (size_t)DIN * SUBD, Zt + 2 * (size_t)NN * SUBD,
                                       Zt16 + 2 * (size_t)NN * SUBD, Z16 + 2 * (size_t)SUBD * NN);

  // 4) Gram partials, M^-1 + logdet, weights, laps
  m_part_kernel<<<dim3(NK, 8), 1024, 0, stream>>>(Zt, Mpart);
  inv_kernel<<<NK, 32, 0, stream>>>(Mpart, Minv, crs, coeff);
  wts_kernel<<<1, 32, 0, stream>>>(crs, lam, wts, laps, out + (size_t)NN * DIN);

  // 5) attention operands
  wt_kernel<<<dim3((NN * SUBD) / 256, NK), 256, 0, stream>>>(Zt, Minv, Wt16);
  g_kernel<<<(NK * DIN * SUBD + 255) / 256, 256, 0, stream>>>(U, Minv, G);

  // 6) masked low-rank flash attention (WMMA QK + PV)
  attn_kernel<<<dim3(NN / 32, NK), 64, 0, stream>>>(Zt16, Wt16, Z16, A16, O);

  // 7) fused epilogue
  final_kernel<<<NN, 128, 0, stream>>>(H, H1, H2, H3, O, G, wts, laps,
                                       thres, ln_gamma, ln_beta, out);
}
